// MixtureOfExperts_29867202576447
// MI455X (gfx1250) — matrix operand order repaired and checked by reading, not compiled, not run on hardware
//
#include <hip/hip_runtime.h>

// ---------------------------------------------------------------------------
// MI455X (gfx1250) MoE block: all GEMM-like stages via v_wmma_f32_16x16x32_bf16
// ---------------------------------------------------------------------------

typedef __attribute__((ext_vector_type(16))) __bf16 v16bf;
typedef __attribute__((ext_vector_type(8)))  __bf16 v8bf;
typedef __attribute__((ext_vector_type(8)))  float  v8f;
typedef __attribute__((ext_vector_type(8)))  unsigned short us8;

#define NTOK   4096      // B*T = 2*2048 tokens
#define TB     2048      // tokens per batch (conv boundary)
#define DIN    512
#define DHID   2048
#define NEXP   10

__device__ __forceinline__ unsigned short f2bf(float f) {
  unsigned u = __float_as_uint(f);
  unsigned r = u + 0x7FFFu + ((u >> 16) & 1u);   // round-to-nearest-even
  return (unsigned short)(r >> 16);
}
__device__ __forceinline__ float bf2f(unsigned short h) {
  return __uint_as_float(((unsigned)h) << 16);
}

__device__ __forceinline__ float act_gelu(float v) {
  return 0.5f * v * (1.0f + erff(v * 0.70710678118654752f));
}
__device__ __forceinline__ float act_silu(float v) {
  return v / (1.0f + __expf(-v));
}
__device__ __forceinline__ float act_elu(float v) {
  return v > 0.0f ? v : (__expf(v) - 1.0f);
}

// ---------------------------------------------------------------------------
// Convert x -> bf16, init residual accumulator z = x, zero the pad buffer.
// ---------------------------------------------------------------------------
__global__ __launch_bounds__(256) void k_convert_x(
    const float* __restrict__ x, unsigned short* __restrict__ xbf,
    float* __restrict__ z, unsigned short* __restrict__ zpad, int n) {
  int i = blockIdx.x * 256 + threadIdx.x;
  if (i < n) { float v = x[i]; xbf[i] = f2bf(v); z[i] = v; }
  if (i < DHID) zpad[i] = 0;   // zero row used for SAME-padding A loads
}

// ---------------------------------------------------------------------------
// Pack one (tap of a) weight matrix into WMMA-B fragment-native bf16 tiles.
// Tile order is (nn, kk) so the K-walk of a fixed 16-column strip is fully
// CONTIGUOUS (1 KB per WMMA step). Within a tile (512 bf16), lane L owns
// bytes [L*32, L*32+32); element j of lane L is
//   B[k = kk*32 + 16*(L>=16) + j, n = nn*16 + (L&15)].
// mode 0: src row-major [K x O] (expert weights)
// mode 1: src conv layout [O x K x taps], take tap dk (conv weights)
// ---------------------------------------------------------------------------
__global__ __launch_bounds__(256) void k_pack_b(
    const float* __restrict__ src, unsigned short* __restrict__ dst,
    int K, int O, int taps, int dk, int mode) {
  int idx = blockIdx.x * 256 + threadIdx.x;
  int total = K * O;
  if (idx >= total) return;
  int numKT = K >> 5;
  int tile = idx >> 9;
  int r    = idx & 511;
  int lane = r >> 4;
  int j    = r & 15;
  int nn = tile / numKT;            // (nn, kk) tile order -> contiguous K-walk
  int kk = tile - nn * numKT;
  int k = (kk << 5) + ((lane >= 16) ? 16 : 0) + j;
  int n = (nn << 4) + (lane & 15);
  float v = (mode == 0) ? src[(long)k * O + n]
                        : src[((long)n * K + k) * taps + dk];
  dst[idx] = f2bf(v);
}

// ---------------------------------------------------------------------------
// Generic taps-accumulated GEMM on WMMA bf16:
//   out[m, n] = epilogue( sum_dk sum_k A[m + dk - pad, k] * B[dk][k, n] + bias[n] )
// A: bf16 row-major [NTOK x K]; rows shifted outside the batch read `zpad`
// (one hoisted 64-bit cndmask per tap instead of per-iteration selects).
// Bp: fragment-packed bf16, taps stacked at stride K*O.
// act: 0 none, 1 GELU, 2 SiLU, 3 ELU
// outmode: 0 store bf16, 1 store f32, 2 f32 +=, 3 f32 += gates[m,e]*val
// Block: 256 thr = 8 waves laid out 2(M) x 4(N); tile = 32 rows x 64 cols.
// ---------------------------------------------------------------------------
__global__ __launch_bounds__(256) void k_conv_gemm_wmma(
    const unsigned short* __restrict__ A,
    const unsigned short* __restrict__ Bp,
    const unsigned short* __restrict__ zpad,
    const float* __restrict__ bias,
    void* __restrict__ Out,
    const float* __restrict__ gates,
    int K, int O, int taps, int pad, int act, int outmode, int expert) {
  const int lane  = threadIdx.x & 31;
  const int wave  = threadIdx.x >> 5;
  const int m0 = blockIdx.y * 32 + (wave >> 2) * 16;
  const int n0 = blockIdx.x * 64 + (wave & 3) * 16;
  if (n0 >= O) return;                     // wave-uniform

  const int batch0 = (m0 / TB) * TB;       // 16-row tiles never cross batches
  const int mrow   = m0 + (lane & 15);     // A-fragment row for this lane
  const int sel8   = (lane >> 4) << 4;     // K-half select (contiguous 16-k half)
  const int numKT  = K >> 5;
  const long tapStride = (long)K * O;      // fragment elements per tap

  v8f acc = (v8f)0.0f;

  for (int dk = 0; dk < taps; ++dk) {
    const int row   = mrow + dk - pad;
    const bool valid = (row >= batch0) && (row < batch0 + TB);
    // Invalid lanes read the 4KB zero buffer: one pointer select per tap,
    // steady-state inner loop is pure global_load_b128 + v_wmma.
    const unsigned short* Arow = valid ? (A + (long)row * K) : zpad;
    // Contiguous B stream for this wave's 16-column strip (1 KB per step).
    const unsigned short* bptr =
        Bp + (long)dk * tapStride + (((long)(n0 >> 4) * numKT) << 9) +
        (lane << 4);
    __builtin_prefetch((const void*)bptr, 0, 1);   // global_prefetch_b8
#pragma unroll 4
    for (int kk = 0; kk < numKT; ++kk) {
      const int k0 = kk << 5;
      // A fragment: lanes 0-15 rows m0..m0+15 / K half 0; lanes 16-31 half 1.
      v8bf lo = __builtin_bit_cast(v8bf, *(const us8*)(Arow + k0 + sel8));
      v8bf hi = __builtin_bit_cast(v8bf, *(const us8*)(Arow + k0 + sel8 + 8));
      v16bf a = __builtin_shufflevector(lo, hi, 0,1,2,3,4,5,6,7,
                                                8,9,10,11,12,13,14,15);
      v16bf b = *(const v16bf*)(bptr + ((long)kk << 9));
      acc = __builtin_amdgcn_wmma_f32_16x16x32_bf16(
          false, a, false, b, (short)0, acc, false, false);
    }
  }

  // Epilogue. C layout: VGPR r -> row m0 + r + 8*(lane>=16), col n0 + lane%16.
  const int col = n0 + (lane & 15);
  const float bn = bias[col];
  const int rowoff = m0 + ((lane >> 4) << 3);
#pragma unroll
  for (int r = 0; r < 8; ++r) {
    const int row = rowoff + r;
    float v = acc[r] + bn;
    if (act == 1)      v = act_gelu(v);
    else if (act == 2) v = act_silu(v);
    else if (act == 3) v = act_elu(v);
    const long o = (long)row * O + col;
    if (outmode == 0)      ((unsigned short*)Out)[o] = f2bf(v);
    else if (outmode == 1) ((float*)Out)[o] = v;
    else if (outmode == 2) ((float*)Out)[o] += v;
    else {
      const float g = gates[row * NEXP + expert];
      ((float*)Out)[o] += g * v;
    }
  }
}

// ---------------------------------------------------------------------------
// Router tail: conv1d(512 -> 10, k=3) on GELU'd rc1, top-2 softmax -> dense
// gates [NTOK x 10] (zeros off the top-k, exactly like the reference).
// One wave per token.
// ---------------------------------------------------------------------------
__global__ __launch_bounds__(128) void k_router(
    const unsigned short* __restrict__ rc1,  // bf16 [NTOK x 512]
    const float* __restrict__ w2,            // [10 x 512 x 3]
    const float* __restrict__ b2,            // [10]
    float* __restrict__ gates) {             // [NTOK x 10]
  const int lane = threadIdx.x & 31;
  const int n = blockIdx.x * 4 + (threadIdx.x >> 5);
  const int batch0 = (n / TB) * TB;
  float part[NEXP];
#pragma unroll
  for (int j = 0; j < NEXP; ++j) part[j] = 0.0f;

  for (int dk = 0; dk < 3; ++dk) {
    const int row = n + dk - 1;
    if (row < batch0 || row >= batch0 + TB) continue;   // wave-uniform
    const unsigned short* xr = rc1 + (long)row * DIN;
    for (int i = lane; i < DIN; i += 32) {
      const float xv = bf2f(xr[i]);
      const float* wp = w2 + (long)i * 3 + dk;
#pragma unroll
      for (int j = 0; j < NEXP; ++j) part[j] += xv * wp[(long)j * DIN * 3];
    }
  }
#pragma unroll
  for (int j = 0; j < NEXP; ++j)
    for (int off = 16; off > 0; off >>= 1)
      part[j] += __shfl_xor(part[j], off, 32);

  if (lane == 0) {
    float lg[NEXP];
#pragma unroll
    for (int j = 0; j < NEXP; ++j) lg[j] = part[j] + b2[j];
    int i0 = 0;
    for (int j = 1; j < NEXP; ++j) if (lg[j] > lg[i0]) i0 = j;
    int i1 = (i0 == 0) ? 1 : 0;
    for (int j = 0; j < NEXP; ++j)
      if (j != i0 && lg[j] > lg[i1]) i1 = j;
    const float g0 = 1.0f / (1.0f + __expf(lg[i1] - lg[i0]));  // softmax top-2
    float* gr = gates + (long)n * NEXP;
#pragma unroll
    for (int j = 0; j < NEXP; ++j) gr[j] = 0.0f;
    gr[i0] = g0;
    gr[i1] = 1.0f - g0;
  }
}

// ---------------------------------------------------------------------------
// LayerNorm over d=512, one wave per token. z already = residual + y + shared.
// ---------------------------------------------------------------------------
__global__ __launch_bounds__(256) void k_layernorm(
    const float* __restrict__ z, const float* __restrict__ g,
    const float* __restrict__ b, float* __restrict__ out) {
  const int lane = threadIdx.x & 31;
  const int n = blockIdx.x * 8 + (threadIdx.x >> 5);
  const float* zr = z + (long)n * DIN;
  float s = 0.0f, s2 = 0.0f;
  for (int i = lane; i < DIN; i += 32) {
    const float v = zr[i];
    s += v; s2 += v * v;
  }
  for (int off = 16; off > 0; off >>= 1) {
    s  += __shfl_xor(s,  off, 32);
    s2 += __shfl_xor(s2, off, 32);
  }
  const float mu  = s * (1.0f / DIN);
  const float var = s2 * (1.0f / DIN) - mu * mu;
  const float inv = rsqrtf(var + 1e-5f);
  float* orow = out + (long)n * DIN;
  for (int i = lane; i < DIN; i += 32)
    orow[i] = (zr[i] - mu) * inv * g[i] + b[i];
}

// ---------------------------------------------------------------------------
// Host orchestration
// ---------------------------------------------------------------------------
extern "C" void kernel_launch(void* const* d_in, const int* in_sizes, int n_in,
                              void* d_out, int out_size, void* d_ws, size_t ws_size,
                              hipStream_t stream) {
  (void)in_sizes; (void)n_in; (void)out_size; (void)ws_size;
  const float* x     = (const float*)d_in[0];
  const float* rc_w1 = (const float*)d_in[1];
  const float* rc_b1 = (const float*)d_in[2];
  const float* rc_w2 = (const float*)d_in[3];
  const float* rc_b2 = (const float*)d_in[4];
  const float* sh_w1 = (const float*)d_in[5];
  const float* sh_b1 = (const float*)d_in[6];
  const float* sh_w2 = (const float*)d_in[7];
  const float* sh_b2 = (const float*)d_in[8];
  const float* e_w1  = (const float*)d_in[9];
  const float* e_b1  = (const float*)d_in[10];
  const float* e_w2  = (const float*)d_in[11];
  const float* e_b2  = (const float*)d_in[12];
  const float* ln_g  = (const float*)d_in[13];
  const float* ln_b  = (const float*)d_in[14];

  // ---- workspace carve (bytes) ----
  char* w = (char*)d_ws;
  auto carve = [&](size_t bytes) {
    char* p = w;
    w += (bytes + 255) & ~(size_t)255;
    return p;
  };
  unsigned short* xbf   = (unsigned short*)carve((size_t)NTOK * DIN * 2);
  unsigned short* rc1   = (unsigned short*)carve((size_t)NTOK * DIN * 2);
  unsigned short* rcw1p = (unsigned short*)carve((size_t)3 * DIN * DIN * 2);
  unsigned short* shw1p = (unsigned short*)carve((size_t)9 * DIN * DHID * 2);
  unsigned short* shw2p = (unsigned short*)carve((size_t)DHID * DIN * 2);
  unsigned short* ew1p  = (unsigned short*)carve((size_t)NEXP * DIN * DHID * 2);
  unsigned short* ew2p  = (unsigned short*)carve((size_t)NEXP * DHID * DIN * 2);
  unsigned short* hbuf  = (unsigned short*)carve((size_t)NTOK * DHID * 2);
  float*          z     = (float*)carve((size_t)NTOK * DIN * 4);
  float*          gates = (float*)carve((size_t)NTOK * NEXP * 4);
  unsigned short* zpad  = (unsigned short*)carve((size_t)DHID * 2);

  // ---- 1) cast x -> bf16, init z = x, zero the SAME-pad row ----
  {
    int n = NTOK * DIN;
    k_convert_x<<<(n + 255) / 256, 256, 0, stream>>>(x, xbf, z, zpad, n);
  }

  // ---- 2) pack all weight matrices into WMMA-B fragment tiles ----
  auto packB = [&](const float* src, unsigned short* dst, int K, int O,
                   int taps, int dk, int mode) {
    int n = K * O;
    k_pack_b<<<(n + 255) / 256, 256, 0, stream>>>(src, dst + (long)dk * K * O,
                                                  K, O, taps, dk, mode);
  };
  for (int dk = 0; dk < 3; ++dk) packB(rc_w1, rcw1p, DIN, DIN, 3, dk, 1);
  for (int dk = 0; dk < 9; ++dk) packB(sh_w1, shw1p, DIN, DHID, 9, dk, 1);
  packB(sh_w2, shw2p, DHID, DIN, 1, 0, 1);
  for (int e = 0; e < NEXP; ++e) {
    packB(e_w1 + (long)e * DIN * DHID, ew1p + (long)e * DIN * DHID,
          DIN, DHID, 1, 0, 0);
    packB(e_w2 + (long)e * DHID * DIN, ew2p + (long)e * DHID * DIN,
          DHID, DIN, 1, 0, 0);
  }

  const dim3 blk(256);
  auto grid_for = [](int O) { return dim3(O / 64, NTOK / 32); };

  // ---- 3) route conv1 (k=3) + GELU -> rc1 (bf16) ----
  k_conv_gemm_wmma<<<grid_for(DIN), blk, 0, stream>>>(
      xbf, rcw1p, zpad, rc_b1, rc1, nullptr,
      DIN, DIN, 3, 1, /*act*/1, /*out*/0, 0);

  // ---- 4) route conv2 + top-2 softmax -> dense gates ----
  k_router<<<NTOK / 4, 128, 0, stream>>>(rc1, rc_w2, rc_b2, gates);

  // ---- 5) shared conv1 (k=9) + SiLU -> hbuf (bf16) ----
  k_conv_gemm_wmma<<<grid_for(DHID), blk, 0, stream>>>(
      xbf, shw1p, zpad, sh_b1, hbuf, nullptr,
      DIN, DHID, 9, 4, /*act*/2, /*out*/0, 0);

  // ---- 6) shared conv2 (k=1): z += hbuf @ W2 + b2 ----
  k_conv_gemm_wmma<<<grid_for(DIN), blk, 0, stream>>>(
      hbuf, shw2p, zpad, sh_b2, z, nullptr,
      DHID, DIN, 1, 0, /*act*/0, /*out*/2, 0);

  // ---- 7) experts (dense over E, matching reference einsum; gates zero
  //         off top-k). hbuf reused after shared branch consumed it. ----
  for (int e = 0; e < NEXP; ++e) {
    k_conv_gemm_wmma<<<grid_for(DHID), blk, 0, stream>>>(
        xbf, ew1p + (long)e * DIN * DHID, zpad, e_b1 + (long)e * DHID, hbuf,
        nullptr, DIN, DHID, 1, 0, /*act ELU*/3, /*out*/0, 0);
    k_conv_gemm_wmma<<<grid_for(DIN), blk, 0, stream>>>(
        hbuf, ew2p + (long)e * DHID * DIN, zpad, e_b2 + (long)e * DIN, z,
        gates, DHID, DIN, 1, 0, /*act*/0, /*out gated +=*/3, e);
  }

  // ---- 8) LayerNorm(z) -> out ----
  k_layernorm<<<NTOK / 8, 256, 0, stream>>>(z, ln_g, ln_b, (float*)d_out);
}
